// CTM_partpad_82815559402223
// MI455X (gfx1250) — matrix operand order repaired and checked by reading, not compiled, not run on hardware
//
#include <hip/hip_runtime.h>

typedef __attribute__((ext_vector_type(16))) __bf16 v16bf;
typedef __attribute__((ext_vector_type(8)))  float  v8f;

constexpr int B_   = 8;
constexpr int N_   = 3136;
constexpr int CIN  = 64;
constexpr int COUT = 128;
constexpr int HG   = 56;
constexpr int WG   = 56;
constexpr int OH   = 28;
constexpr int OW   = 28;
constexpr int M_   = 784;
constexpr int SB   = 24;   // LDS B-tile row stride in bf16 (48B, 16B-aligned)

#define DEVI __device__ __forceinline__

// ---------------- WMMA helpers (CDNA5 layouts, 16x16x32 bf16) ----------------

DEVI v8f wmma_bf16(v16bf a, v16bf b, v8f c) {
  return __builtin_amdgcn_wmma_f32_16x16x32_bf16(false, a, false, b, (short)0, c,
                                                 false, false);
}

// A fragment: 16x32 bf16. lane l: row = l&15, h = l>>4.
// VGPR v holds K pair: kbase = (v<4 ? 2v : 16+2(v-4)) + 8h (+ 32*kk chunk).
DEVI void load_afrag(const __bf16* rowPtr, int h, v16bf (&af)[4]) {
#pragma unroll
  for (int kk = 0; kk < 4; ++kk) {
#pragma unroll
    for (int v = 0; v < 8; ++v) {
      int kb = kk * 32 + ((v < 4) ? (2 * v) : (16 + 2 * (v - 4))) + 8 * h;
      union { unsigned u; __bf16 e[2]; } cv;
      cv.u = *reinterpret_cast<const unsigned*>(rowPtr + kb);
      af[kk][2 * v]     = cv.e[0];
      af[kk][2 * v + 1] = cv.e[1];
    }
  }
}

// Stage a 16x128 bf16 tile (rows contiguous in global) into LDS TRANSPOSED:
// ldsBt[k * SB + n] = gsrc[n * 128 + k].  Each thread: 8 x b32 global loads
// (two k-columns over 8 rows) -> 2 x ds_store_b128.
DEVI void stage_btile(const __bf16* __restrict__ gsrc, __bf16* ldsBt, int tid) {
  int n0 = (tid & 1) * 8;
  int k0 = tid & ~1;
  union { unsigned u; __bf16 e[2]; } ld[8];
#pragma unroll
  for (int r = 0; r < 8; ++r)
    ld[r].u = *reinterpret_cast<const unsigned*>(gsrc + (size_t)(n0 + r) * COUT + k0);
  union { uint4 q; __bf16 e[8]; } s0, s1;
#pragma unroll
  for (int r = 0; r < 8; ++r) { s0.e[r] = ld[r].e[0]; s1.e[r] = ld[r].e[1]; }
  *reinterpret_cast<uint4*>(ldsBt + (size_t)k0 * SB + n0)       = s0.q;
  *reinterpret_cast<uint4*>(ldsBt + (size_t)(k0 + 1) * SB + n0) = s1.q;
}

// B fragment: 32x16 bf16 from transposed LDS tile. lane l holds column n = l&15;
// element s holds k = kk*32 + 8h + (s<8 ? s : s+8) (same order as A).
DEVI v16bf load_bfrag_t(const __bf16* ldsBt, int kk, int lane) {
  const __bf16* p = ldsBt + (size_t)(kk * 32) * SB + (lane & 15);
  int h = lane >> 4;
  v16bf v;
#pragma unroll
  for (int s = 0; s < 16; ++s) v[s] = p[(size_t)((s < 8 ? s : s + 8) + 8 * h) * SB];
  return v;
}

DEVI void ins5(float (&t)[5], float d) {
  if (d < t[4]) {
    t[4] = d;
#pragma unroll
    for (int k = 4; k > 0; --k)
      if (t[k] < t[k - 1]) { float tmp = t[k]; t[k] = t[k - 1]; t[k - 1] = tmp; }
  }
}

// ---------------- generic zero ----------------
__global__ void k_zero(float* p, int n) {
  int i = blockIdx.x * blockDim.x + threadIdx.x;
  if (i < n) p[i] = 0.f;
}

// ---------------- token2map scatter ----------------
__global__ void k_t2m(const float* __restrict__ x, const float* __restrict__ loc,
                      const int* __restrict__ idx_agg, float* grid, float* cnt) {
  int bn = blockIdx.x;            // b*N + n
  int c = threadIdx.x;            // 0..63
  int b = bn / N_;
  float lx = fminf(fmaxf(loc[(size_t)bn * 2 + 0], -1.f), 1.f);
  float ly = fminf(fmaxf(loc[(size_t)bn * 2 + 1], -1.f), 1.f);
  int xi = (int)rintf((lx + 1.f) * 0.5f * (WG - 1));
  int yi = (int)rintf((ly + 1.f) * 0.5f * (HG - 1));
  int hw = yi * WG + xi;
  int ia = idx_agg[bn];
  float v = x[((size_t)b * N_ + ia) * CIN + c];
  atomicAdd(&grid[((size_t)b * (HG * WG) + hw) * CIN + c], v);
  if (c == 0) atomicAdd(&cnt[(size_t)b * (HG * WG) + hw], 1.0f);
}

__global__ void k_gridnorm(float* grid, const float* cnt, int total) {
  int i = blockIdx.x * blockDim.x + threadIdx.x;
  if (i < total) grid[i] /= fmaxf(cnt[i / CIN], 1e-6f);
}

// ---------------- 3x3 stride-2 conv (NHWC, HWIO) ----------------
__global__ void k_conv(const float* __restrict__ grid, const float* __restrict__ cw,
                       const float* __restrict__ cb, float* __restrict__ out) {
  __shared__ float patch[9][CIN];
  int p = blockIdx.x;                        // b*784 + oy*28 + ox
  int co = threadIdx.x;                      // 0..127
  int b = p / (OH * OW), rem = p % (OH * OW);
  int oy = rem / OW, ox = rem % OW;
  for (int idx = threadIdx.x; idx < 9 * CIN; idx += COUT) {
    int kyx = idx / CIN, ci = idx % CIN;
    int iy = 2 * oy - 1 + kyx / 3, ix = 2 * ox - 1 + kyx % 3;
    patch[kyx][ci] = (iy >= 0 && iy < HG && ix >= 0 && ix < WG)
                         ? grid[((size_t)b * (HG * WG) + iy * WG + ix) * CIN + ci]
                         : 0.f;
  }
  __syncthreads();
  float acc = cb[co];
  for (int kyx = 0; kyx < 9; ++kyx)
#pragma unroll 8
    for (int ci = 0; ci < CIN; ++ci)
      acc += patch[kyx][ci] * cw[(kyx * CIN + ci) * COUT + co];
  out[(size_t)p * COUT + co] = acc;
}

// ---------------- map2token scatter ----------------
__global__ void k_m2t(const float* __restrict__ conv, const float* __restrict__ loc,
                      const int* __restrict__ idx_agg, const float* __restrict__ aw,
                      float* sv, float* sw) {
  int bn = blockIdx.x;
  int c = threadIdx.x;                       // 0..127
  int b = bn / N_;
  float lx = fminf(fmaxf(loc[(size_t)bn * 2 + 0], -1.f), 1.f);
  float ly = fminf(fmaxf(loc[(size_t)bn * 2 + 1], -1.f), 1.f);
  int xi = (int)rintf((lx + 1.f) * 0.5f * (OW - 1));
  int yi = (int)rintf((ly + 1.f) * 0.5f * (OH - 1));
  int hw = yi * OW + xi;
  int ia = idx_agg[bn];
  float w = aw[bn];
  float v = conv[((size_t)b * (OH * OW) + hw) * COUT + c];
  atomicAdd(&sv[((size_t)b * N_ + ia) * COUT + c], v * w);
  if (c == 0) atomicAdd(&sw[(size_t)b * N_ + ia], w);
}

// ---------------- per-token: skip GEMM + LN + conf; emit f32/bf16/sq/weight ----
__global__ void k_tok(const float* __restrict__ x, const float* __restrict__ skw,
                      const float* __restrict__ sv, const float* __restrict__ sw,
                      const float* __restrict__ lng, const float* __restrict__ lnb,
                      const float* __restrict__ cfw, const float* __restrict__ cfb,
                      float* x_tok, __bf16* Xb, float* sq, float* wgt) {
  __shared__ float xrow[CIN];
  __shared__ float red[COUT];
  int bn = blockIdx.x;
  int c = threadIdx.x;
  if (c < CIN) xrow[c] = x[(size_t)bn * CIN + c];
  __syncthreads();
  float acc = sv[(size_t)bn * COUT + c] / fmaxf(sw[bn], 1e-6f);
  float s = 0.f;
#pragma unroll 8
  for (int k = 0; k < CIN; ++k) s += xrow[k] * skw[k * COUT + c];
  float xt = acc + s;

  red[c] = xt; __syncthreads();
  for (int st = 64; st > 0; st >>= 1) { if (c < st) red[c] += red[c + st]; __syncthreads(); }
  float mu = red[0] / COUT; __syncthreads();
  float d = xt - mu;
  red[c] = d * d; __syncthreads();
  for (int st = 64; st > 0; st >>= 1) { if (c < st) red[c] += red[c + st]; __syncthreads(); }
  float var = red[0] / COUT; __syncthreads();
  float xln = d * rsqrtf(var + 1e-5f) * lng[c] + lnb[c];

  x_tok[(size_t)bn * COUT + c] = xln;
  Xb[(size_t)bn * COUT + c] = (__bf16)xln;

  red[c] = xln * xln; __syncthreads();
  for (int st = 64; st > 0; st >>= 1) { if (c < st) red[c] += red[c + st]; __syncthreads(); }
  if (c == 0) sq[bn] = red[0];
  __syncthreads();
  red[c] = xln * cfw[c]; __syncthreads();
  for (int st = 64; st > 0; st >>= 1) { if (c < st) red[c] += red[c + st]; __syncthreads(); }
  if (c == 0) wgt[bn] = expf(red[0] + cfb[0]);
}

// ---------------- pass A: WMMA Gram tiles -> kNN density + dmax ----------------
__global__ void k_dens(const __bf16* __restrict__ Xb, const float* __restrict__ sq,
                       float* dens, unsigned* dmaxbits) {
  __shared__ __align__(16) __bf16 ldsBt[128 * SB];
  __shared__ float ldsD[4][16][17];
  __shared__ float sqi[64];
  __shared__ float sqj[16];
  int b = blockIdx.y;
  int rowBase = blockIdx.x * 64;
  int tid = threadIdx.x, w = tid >> 5, lane = tid & 31, h = lane >> 4;
  if (tid < 64) sqi[tid] = sq[(size_t)b * N_ + rowBase + tid];

  const __bf16* Arow = Xb + ((size_t)b * N_ + rowBase + w * 16 + (lane & 15)) * COUT;
  v16bf af[4];
  load_afrag(Arow, h, af);

  float t5[5] = {3.4e38f, 3.4e38f, 3.4e38f, 3.4e38f, 3.4e38f};
  float dmx = 0.f;

  for (int jt = 0; jt < N_ / 16; ++jt) {
    int j0 = jt * 16;
    __syncthreads();
    stage_btile(Xb + ((size_t)b * N_ + j0) * COUT, ldsBt, tid);
    if (tid < 16) sqj[tid] = sq[(size_t)b * N_ + j0 + tid];
    __syncthreads();
    v8f acc = {};
#pragma unroll
    for (int kk = 0; kk < 4; ++kk) acc = wmma_bf16(af[kk], load_bfrag_t(ldsBt, kk, lane), acc);
    int col = lane & 15;
#pragma unroll
    for (int r = 0; r < 8; ++r) {
      int row = r + 8 * h;
      ldsD[w][row][col] = fmaxf(sqi[w * 16 + row] + sqj[col] - 2.0f * acc[r], 0.f);
    }
    __syncthreads();
    if (lane < 16) {
      for (int nn = 0; nn < 16; ++nn) {
        float dd = ldsD[w][lane][nn];
        dmx = fmaxf(dmx, dd);
        ins5(t5, dd);
      }
    }
  }
  if (lane < 16) {
    float mean = (t5[0] + t5[1] + t5[2] + t5[3] + t5[4]) * 0.2f;
    dens[(size_t)b * N_ + rowBase + w * 16 + lane] = expf(-mean);
    atomicMax(&dmaxbits[b], __float_as_uint(dmx));
  }
}

// ---------------- pass B: recompute tiles -> delta, score -------------------
__global__ void k_score(const __bf16* __restrict__ Xb, const float* __restrict__ sq,
                        const float* __restrict__ dens, const unsigned* __restrict__ dmaxbits,
                        float* score) {
  __shared__ __align__(16) __bf16 ldsBt[128 * SB];
  __shared__ float ldsD[4][16][17];
  __shared__ float sqi[64];
  __shared__ float sqj[16];
  __shared__ float dnj[16];
  int b = blockIdx.y;
  int rowBase = blockIdx.x * 64;
  int tid = threadIdx.x, w = tid >> 5, lane = tid & 31, h = lane >> 4;
  if (tid < 64) sqi[tid] = sq[(size_t)b * N_ + rowBase + tid];
  float dmx = __uint_as_float(dmaxbits[b]);

  const __bf16* Arow = Xb + ((size_t)b * N_ + rowBase + w * 16 + (lane & 15)) * COUT;
  v16bf af[4];
  load_afrag(Arow, h, af);

  int myrow = rowBase + w * 16 + (lane & 15);
  float dens_i = dens[(size_t)b * N_ + myrow];
  float delta = dmx;

  for (int jt = 0; jt < N_ / 16; ++jt) {
    int j0 = jt * 16;
    __syncthreads();
    stage_btile(Xb + ((size_t)b * N_ + j0) * COUT, ldsBt, tid);
    if (tid < 16) {
      sqj[tid] = sq[(size_t)b * N_ + j0 + tid];
      dnj[tid] = dens[(size_t)b * N_ + j0 + tid];
    }
    __syncthreads();
    v8f acc = {};
#pragma unroll
    for (int kk = 0; kk < 4; ++kk) acc = wmma_bf16(af[kk], load_bfrag_t(ldsBt, kk, lane), acc);
    int col = lane & 15;
#pragma unroll
    for (int r = 0; r < 8; ++r) {
      int row = r + 8 * h;
      ldsD[w][row][col] = fmaxf(sqi[w * 16 + row] + sqj[col] - 2.0f * acc[r], 0.f);
    }
    __syncthreads();
    if (lane < 16) {
      for (int nn = 0; nn < 16; ++nn) {
        float dd = ldsD[w][lane][nn];
        if (dnj[nn] > dens_i) delta = fminf(delta, dd);
      }
    }
  }
  if (lane < 16) score[(size_t)b * N_ + myrow] = delta * dens_i;
}

// ---------------- pass C: per-batch top-M via bitonic sort ------------------
__global__ void k_topm(const float* __restrict__ score, int* index_down) {
  __shared__ float key[4096];
  __shared__ int kid[4096];
  int b = blockIdx.x, tid = threadIdx.x;  // 1024 threads
  for (int s = tid; s < 4096; s += 1024) {
    key[s] = (s < N_) ? score[(size_t)b * N_ + s] : -3.4e38f;
    kid[s] = s;
  }
  __syncthreads();
  for (int k = 2; k <= 4096; k <<= 1) {
    for (int j = k >> 1; j > 0; j >>= 1) {
      for (int t = tid; t < 2048; t += 1024) {
        int i1 = 2 * j * (t / j) + (t % j);
        int i2 = i1 + j;
        bool desc = ((i1 & k) == 0);
        float a = key[i1], c = key[i2];
        bool sw = desc ? (a < c) : (a > c);
        if (sw) {
          key[i1] = c; key[i2] = a;
          int tm = kid[i1]; kid[i1] = kid[i2]; kid[i2] = tm;
        }
      }
      __syncthreads();
    }
  }
  for (int m = tid; m < M_; m += 1024) index_down[b * M_ + m] = kid[m];
}

// ---------------- gather cluster-center features ----------------------------
__global__ void k_gatherc(const __bf16* __restrict__ Xb, const float* __restrict__ sq,
                          const int* __restrict__ index_down, __bf16* Xc, float* sqc) {
  int bm = blockIdx.x;            // b*M + m
  int c = threadIdx.x;            // 0..127
  int b = bm / M_;
  int id = index_down[bm];
  Xc[(size_t)bm * COUT + c] = Xb[((size_t)b * N_ + id) * COUT + c];
  if (c == 0) sqc[bm] = sq[(size_t)b * N_ + id];
}

// ---------------- pass D: WMMA assignment (argmin over centers) -------------
__global__ void k_assign(const __bf16* __restrict__ Xb, const __bf16* __restrict__ Xc,
                         const float* __restrict__ sq, const float* __restrict__ sqc,
                         int* idx_cluster) {
  __shared__ __align__(16) __bf16 ldsBt[128 * SB];
  __shared__ float ldsD[4][16][17];
  __shared__ float sqi[64];
  __shared__ float sqj[16];
  int b = blockIdx.y;
  int rowBase = blockIdx.x * 64;
  int tid = threadIdx.x, w = tid >> 5, lane = tid & 31, h = lane >> 4;
  if (tid < 64) sqi[tid] = sq[(size_t)b * N_ + rowBase + tid];

  const __bf16* Arow = Xb + ((size_t)b * N_ + rowBase + w * 16 + (lane & 15)) * COUT;
  v16bf af[4];
  load_afrag(Arow, h, af);

  float best = 3.4e38f;
  int bestm = 0;

  for (int ct = 0; ct < M_ / 16; ++ct) {
    int j0 = ct * 16;
    __syncthreads();
    stage_btile(Xc + ((size_t)b * M_ + j0) * COUT, ldsBt, tid);
    if (tid < 16) sqj[tid] = sqc[(size_t)b * M_ + j0 + tid];
    __syncthreads();
    v8f acc = {};
#pragma unroll
    for (int kk = 0; kk < 4; ++kk) acc = wmma_bf16(af[kk], load_bfrag_t(ldsBt, kk, lane), acc);
    int col = lane & 15;
#pragma unroll
    for (int r = 0; r < 8; ++r) {
      int row = r + 8 * h;
      ldsD[w][row][col] = fmaxf(sqi[w * 16 + row] + sqj[col] - 2.0f * acc[r], 0.f);
    }
    __syncthreads();
    if (lane < 16) {
      for (int nn = 0; nn < 16; ++nn) {
        float dd = ldsD[w][lane][nn];
        int m = j0 + nn;
        if (dd < best) { best = dd; bestm = m; }
      }
    }
  }
  if (lane < 16)
    idx_cluster[(size_t)b * N_ + rowBase + w * 16 + (lane & 15)] = bestm;
}

// ---------------- merge ----------------
__global__ void k_merge1(const int* __restrict__ idx_cluster, const float* __restrict__ wgt,
                         float* allw) {
  int tid = blockIdx.x * blockDim.x + threadIdx.x;
  if (tid >= B_ * N_) return;
  int b = tid / N_;
  atomicAdd(&allw[b * M_ + idx_cluster[tid]], wgt[tid]);
}

__global__ void k_merge2(const int* __restrict__ idx_cluster, const float* __restrict__ wgt,
                         const float* __restrict__ allw, const float* __restrict__ x_tok,
                         float* x_down, float* norm_w) {
  int bn = blockIdx.x;
  int c = threadIdx.x;            // 0..127
  int b = bn / N_;
  int ic = idx_cluster[bn];
  float nw = wgt[bn] / (allw[b * M_ + ic] + 1e-6f);
  if (c == 0) norm_w[bn] = nw;
  atomicAdd(&x_down[((size_t)b * M_ + ic) * COUT + c], x_tok[(size_t)bn * COUT + c] * nw);
}

// ---------------- final outputs ----------------
__global__ void k_final1(const int* __restrict__ idx_agg, const int* __restrict__ idx_cluster,
                         const float* __restrict__ aw, const float* __restrict__ norm_w,
                         float* out_idx, float* awd, unsigned* maxbits) {
  int tid = blockIdx.x * blockDim.x + threadIdx.x;
  if (tid >= B_ * N_) return;
  int b = tid / N_;
  int ia = idx_agg[tid];
  int ic = idx_cluster[(size_t)b * N_ + ia];
  out_idx[tid] = (float)ic;
  float v = aw[tid] * norm_w[(size_t)b * N_ + ia];
  awd[tid] = v;
  atomicMax(&maxbits[b], __float_as_uint(v));
}

__global__ void k_final2(const float* __restrict__ awd, const unsigned* __restrict__ maxbits,
                         float* out_awd) {
  int tid = blockIdx.x * blockDim.x + threadIdx.x;
  if (tid >= B_ * N_) return;
  int b = tid / N_;
  out_awd[tid] = awd[tid] / __uint_as_float(maxbits[b]);
}

// =============================== host launch ===============================
extern "C" void kernel_launch(void* const* d_in, const int* in_sizes, int n_in,
                              void* d_out, int out_size, void* d_ws, size_t ws_size,
                              hipStream_t stream) {
  const float* x       = (const float*)d_in[0];
  const float* loc     = (const float*)d_in[1];
  const float* aw      = (const float*)d_in[2];
  const float* conv_w  = (const float*)d_in[3];
  const float* conv_b  = (const float*)d_in[4];
  const float* skip_w  = (const float*)d_in[5];
  const float* ln_g    = (const float*)d_in[6];
  const float* ln_b    = (const float*)d_in[7];
  const float* conf_w  = (const float*)d_in[8];
  const float* conf_b  = (const float*)d_in[9];
  const int*   idx_agg = (const int*)d_in[10];
  // d_in[11..13] (idx_k_loc, H, W) unused: constants baked in.

  char* w = (char*)d_ws;
  auto nxt = [&](size_t bytes) -> void* {
    void* p = (void*)w;
    w += (bytes + 255) & ~(size_t)255;
    return p;
  };
  float*    grid     = (float*)nxt((size_t)B_ * HG * WG * CIN * 4);
  float*    cnt      = (float*)nxt((size_t)B_ * HG * WG * 4);
  float*    convout  = (float*)nxt((size_t)B_ * OH * OW * COUT * 4);
  float*    sv       = (float*)nxt((size_t)B_ * N_ * COUT * 4);
  float*    sw       = (float*)nxt((size_t)B_ * N_ * 4);
  float*    x_tok    = (float*)nxt((size_t)B_ * N_ * COUT * 4);
  __bf16*   Xb       = (__bf16*)nxt((size_t)B_ * N_ * COUT * 2);
  float*    sq       = (float*)nxt((size_t)B_ * N_ * 4);
  float*    wgt      = (float*)nxt((size_t)B_ * N_ * 4);
  float*    dens     = (float*)nxt((size_t)B_ * N_ * 4);
  unsigned* dmaxbits = (unsigned*)nxt((size_t)B_ * 4);
  float*    score    = (float*)nxt((size_t)B_ * N_ * 4);
  int*      indexdn  = (int*)nxt((size_t)B_ * M_ * 4);
  __bf16*   Xc       = (__bf16*)nxt((size_t)B_ * M_ * COUT * 2);
  float*    sqc      = (float*)nxt((size_t)B_ * M_ * 4);
  int*      idx_cl   = (int*)nxt((size_t)B_ * N_ * 4);
  float*    allw     = (float*)nxt((size_t)B_ * M_ * 4);
  float*    norm_w   = (float*)nxt((size_t)B_ * N_ * 4);
  float*    awd      = (float*)nxt((size_t)B_ * N_ * 4);
  unsigned* maxbits  = (unsigned*)nxt((size_t)B_ * 4);

  float* out_xdown = (float*)d_out;                              // B*M*COUT
  float* out_idx   = out_xdown + (size_t)B_ * M_ * COUT;         // B*N
  float* out_awd   = out_idx + (size_t)B_ * N_;                  // B*N

  auto zero = [&](float* p, size_t n) {
    k_zero<<<(unsigned)((n + 255) / 256), 256, 0, stream>>>(p, (int)n);
  };
  zero(grid, (size_t)B_ * HG * WG * CIN);
  zero(cnt, (size_t)B_ * HG * WG);
  zero(sv, (size_t)B_ * N_ * COUT);
  zero(sw, (size_t)B_ * N_);
  zero((float*)dmaxbits, B_);
  zero(allw, (size_t)B_ * M_);
  zero((float*)maxbits, B_);
  zero(out_xdown, (size_t)B_ * M_ * COUT);

  k_t2m<<<B_ * N_, CIN, 0, stream>>>(x, loc, idx_agg, grid, cnt);
  k_gridnorm<<<(B_ * HG * WG * CIN + 255) / 256, 256, 0, stream>>>(
      grid, cnt, B_ * HG * WG * CIN);
  k_conv<<<B_ * OH * OW, COUT, 0, stream>>>(grid, conv_w, conv_b, convout);
  k_m2t<<<B_ * N_, COUT, 0, stream>>>(convout, loc, idx_agg, aw, sv, sw);
  k_tok<<<B_ * N_, COUT, 0, stream>>>(x, skip_w, sv, sw, ln_g, ln_b, conf_w, conf_b,
                                      x_tok, Xb, sq, wgt);

  dim3 gtiles(N_ / 64, B_);
  k_dens<<<gtiles, 128, 0, stream>>>(Xb, sq, dens, dmaxbits);
  k_score<<<gtiles, 128, 0, stream>>>(Xb, sq, dens, dmaxbits, score);
  k_topm<<<B_, 1024, 0, stream>>>(score, indexdn);
  k_gatherc<<<B_ * M_, COUT, 0, stream>>>(Xb, sq, indexdn, Xc, sqc);
  k_assign<<<gtiles, 128, 0, stream>>>(Xb, Xc, sq, sqc, idx_cl);

  k_merge1<<<(B_ * N_ + 255) / 256, 256, 0, stream>>>(idx_cl, wgt, allw);
  k_merge2<<<B_ * N_, COUT, 0, stream>>>(idx_cl, wgt, allw, x_tok, out_xdown, norm_w);
  k_final1<<<(B_ * N_ + 255) / 256, 256, 0, stream>>>(idx_agg, idx_cl, aw, norm_w,
                                                      out_idx, awd, maxbits);
  k_final2<<<(B_ * N_ + 255) / 256, 256, 0, stream>>>(awd, maxbits, out_awd);
}
